// FastMoELayer_59313498358379
// MI455X (gfx1250) — matrix operand order repaired and checked by reading, not compiled, not run on hardware
//
#include <hip/hip_runtime.h>
#include <cstdint>

// Problem constants (from reference setup_inputs)
#define T_TOK 8192   // B*S = 4*2048
#define DDIM  2048
#define HDIM  4096
#define ODIM  2048
#define NEXP  8

typedef __attribute__((ext_vector_type(16))) __bf16 v16bf;
typedef __attribute__((ext_vector_type(8)))  float  v8f;

struct FragBits { uint4 lo, hi; };   // 32 bytes -> bit_cast to v16bf

static __device__ __forceinline__ unsigned short f32_to_bf16_rte(float f) {
  unsigned int u = __float_as_uint(f);
  u += 0x7FFFu + ((u >> 16) & 1u);   // round-to-nearest-even
  return (unsigned short)(u >> 16);
}

static __device__ __forceinline__ float silu_fast(float v) {
  // v * rcp(1 + exp(-v)) : v_exp_f32 + v_rcp_f32, no IEEE divide sequence
  return v * __builtin_amdgcn_rcpf(1.f + __expf(-v));
}

// ---------------------------------------------------------------- utilities
__global__ void zero_i32_kernel(int* __restrict__ p, int n) {
  int i = blockIdx.x * blockDim.x + threadIdx.x;
  if (i < n) p[i] = 0;
}

__global__ void zero_f32_kernel(float* __restrict__ p, size_t n4) {
  size_t i = (size_t)blockIdx.x * blockDim.x + threadIdx.x;
  size_t stride = (size_t)gridDim.x * blockDim.x;
  float4 z; z.x = 0.f; z.y = 0.f; z.z = 0.f; z.w = 0.f;
  for (; i < n4; i += stride) ((float4*)p)[i] = z;
}

__global__ void cvt_f32_bf16_kernel(const float* __restrict__ src,
                                    unsigned short* __restrict__ dst,
                                    size_t n4) {
  size_t i = (size_t)blockIdx.x * blockDim.x + threadIdx.x;
  size_t stride = (size_t)gridDim.x * blockDim.x;
  for (; i < n4; i += stride) {
    float4 v = ((const float4*)src)[i];
    unsigned int a = f32_to_bf16_rte(v.x);
    unsigned int b = f32_to_bf16_rte(v.y);
    unsigned int c = f32_to_bf16_rte(v.z);
    unsigned int d = f32_to_bf16_rte(v.w);
    uint2 o; o.x = a | (b << 16); o.y = c | (d << 16);
    ((uint2*)dst)[i] = o;
  }
}

// ---------------------------------------------------------------- gating
__global__ __launch_bounds__(256)
void moe_gate_kernel(const float* __restrict__ x,
                     const float* __restrict__ Wg,
                     const float* __restrict__ bg,
                     float* __restrict__ gates,
                     int* __restrict__ counts,
                     int* __restrict__ lists) {
  const int wid  = threadIdx.x >> 5;
  const int lane = threadIdx.x & 31;
  const int t = blockIdx.x * 8 + wid;
  if (t >= T_TOK) return;

  float acc[NEXP];
#pragma unroll
  for (int e = 0; e < NEXP; ++e) acc[e] = 0.f;

  const float* xr = x + (size_t)t * DDIM;
  for (int d = lane; d < DDIM; d += 32) {
    float xv = xr[d];
#pragma unroll
    for (int e = 0; e < NEXP; ++e) acc[e] += xv * Wg[e * DDIM + d];
  }
#pragma unroll
  for (int e = 0; e < NEXP; ++e) {
#pragma unroll
    for (int off = 16; off > 0; off >>= 1)
      acc[e] += __shfl_xor(acc[e], off, 32);
  }

  if (lane == 0) {
    float sc[NEXP];
    float mx = -1e30f;
#pragma unroll
    for (int e = 0; e < NEXP; ++e) { sc[e] = acc[e] + bg[e]; mx = fmaxf(mx, sc[e]); }
    float sum = 0.f;
#pragma unroll
    for (int e = 0; e < NEXP; ++e) { sc[e] = __expf(sc[e] - mx); sum += sc[e]; }
    float inv = 1.f / sum;
#pragma unroll
    for (int e = 0; e < NEXP; ++e) sc[e] *= inv;

    int i1 = 0;
#pragma unroll
    for (int e = 1; e < NEXP; ++e) if (sc[e] > sc[i1]) i1 = e;
    int i2 = (i1 == 0) ? 1 : 0;
#pragma unroll
    for (int e = 0; e < NEXP; ++e) if (e != i1 && sc[e] > sc[i2]) i2 = e;

    float s = fmaxf(sc[i1] + sc[i2], 1e-12f);
    float* grow = gates + (size_t)t * NEXP;
#pragma unroll
    for (int e = 0; e < NEXP; ++e) grow[e] = 0.f;
    grow[i1] = sc[i1] / s;
    grow[i2] = sc[i2] / s;

    int p1 = atomicAdd(&counts[i1], 1);
    lists[i1 * T_TOK + p1] = t;
    int p2 = atomicAdd(&counts[i2], 1);
    lists[i2 * T_TOK + p2] = t;
  }
}

// ---------------------------------------------------------------- GEMM core
// Block: 256 threads = 8 waves. Block tile M=128 x N=128.
// Wave (wm: 32-row band, wn: 64-col band) owns 32x64 = 8 WMMA accumulators.
// Double-buffered LDS: one barrier per 32-wide K-step; ds_stores of tile k+1
// overlap the WMMA chain of tile k.
#define ST 40   // padded LDS row stride in shorts (32 K-shorts + 8 pad)

static __device__ __forceinline__ void wave_mma_step(
    const unsigned short* sAbuf, const unsigned short* sBbuf,
    int wm, int wn, int lane, v8f acc[2][4]) {
  v16bf afr[2];
#pragma unroll
  for (int i = 0; i < 2; ++i) {
    // A layout: lane l holds row (l&15); K = (l>>4)*8 + [0..7], +16 for hi.
    FragBits fa;
    const int ao = (wm * 32 + i * 16 + (lane & 15)) * ST + (lane >> 4) * 8;
    fa.lo = *(const uint4*)&sAbuf[ao];
    fa.hi = *(const uint4*)&sAbuf[ao + 16];
    afr[i] = __builtin_bit_cast(v16bf, fa);
  }
  v16bf bfr[4];
#pragma unroll
  for (int j = 0; j < 4; ++j) {
    // B layout: lane l holds col (l&15); K = (l>>4)*8 + [0..7], +16 for hi.
    FragBits fb;
    const int bo = (wn * 64 + j * 16 + (lane & 15)) * ST + (lane >> 4) * 8;
    fb.lo = *(const uint4*)&sBbuf[bo];
    fb.hi = *(const uint4*)&sBbuf[bo + 16];
    bfr[j] = __builtin_bit_cast(v16bf, fb);
  }
#pragma unroll
  for (int i = 0; i < 2; ++i)
#pragma unroll
    for (int j = 0; j < 4; ++j)
      acc[i][j] = __builtin_amdgcn_wmma_f32_16x16x32_bf16(
          false, afr[i], false, bfr[j], (short)0, acc[i][j], false, false);
}

// h = silu(x_gathered @ Wh[e]^T + bh[e]) stored compacted by list position.
__global__ __launch_bounds__(256)
void moe_ffn1_kernel(const unsigned short* __restrict__ xb,
                     const unsigned short* __restrict__ Whb,
                     const float* __restrict__ bh,
                     const int* __restrict__ counts,
                     const int* __restrict__ lists,
                     unsigned short* __restrict__ h,
                     int e) {
  const int cnt = counts[e];
  if ((int)(blockIdx.y * 128) >= cnt) return;

  __shared__ unsigned short sA[2][128 * ST];
  __shared__ unsigned short sB[2][128 * ST];
  __shared__ int sTok[128];

  const int tid  = threadIdx.x;
  const int lane = tid & 31;
  const int wid  = tid >> 5;
  const int wm   = wid & 3;   // 32-row band
  const int wn   = wid >> 2;  // 64-col band

  const int* lst = lists + e * T_TOK;
  {
    int idx0 = (int)blockIdx.y * 128 + (tid & 127);
    if (idx0 >= cnt) idx0 = cnt - 1;      // cnt >= 1 guaranteed here
    if (tid < 128) sTok[tid] = lst[idx0];
  }
  __syncthreads();

  const int arow = tid >> 2;            // 0..63
  const int aseg = (tid & 3) * 8;       // short offset in 32-wide K tile
  const int sOffLo = arow * ST + aseg;
  const int sOffHi = (arow + 64) * ST + aseg;
  const unsigned short* Arow0 = xb + (size_t)sTok[arow] * DDIM + aseg;
  const unsigned short* Arow1 = xb + (size_t)sTok[arow + 64] * DDIM + aseg;
  const unsigned short* Brow0 = Whb + (size_t)e * HDIM * DDIM
                              + (size_t)((int)blockIdx.x * 128 + arow) * DDIM + aseg;
  const unsigned short* Brow1 = Brow0 + (size_t)64 * DDIM;

  v8f acc[2][4];
#pragma unroll
  for (int i = 0; i < 2; ++i)
#pragma unroll
    for (int j = 0; j < 4; ++j)
#pragma unroll
      for (int r = 0; r < 8; ++r) acc[i][j][r] = 0.f;

  // Prologue: stage K-tile 0 into buffer 0.
  {
    uint4 a0 = *(const uint4*)(Arow0);
    uint4 a1 = *(const uint4*)(Arow1);
    uint4 b0 = *(const uint4*)(Brow0);
    uint4 b1 = *(const uint4*)(Brow1);
    *(uint4*)&sA[0][sOffLo] = a0;
    *(uint4*)&sA[0][sOffHi] = a1;
    *(uint4*)&sB[0][sOffLo] = b0;
    *(uint4*)&sB[0][sOffHi] = b1;
  }
  __syncthreads();

  // DDIM is a multiple of 64: process two 32-K stages per trip (ping-pong).
  for (int k0 = 0; k0 < DDIM; k0 += 64) {
    { // stage p=0 holds k0; prefetch k0+32 into p=1 while computing
      const int kn = k0 + 32;
      uint4 a0 = *(const uint4*)(Arow0 + kn);
      uint4 a1 = *(const uint4*)(Arow1 + kn);
      uint4 b0 = *(const uint4*)(Brow0 + kn);
      uint4 b1 = *(const uint4*)(Brow1 + kn);
      wave_mma_step(sA[0], sB[0], wm, wn, lane, acc);
      *(uint4*)&sA[1][sOffLo] = a0;
      *(uint4*)&sA[1][sOffHi] = a1;
      *(uint4*)&sB[1][sOffLo] = b0;
      *(uint4*)&sB[1][sOffHi] = b1;
      __syncthreads();
    }
    { // stage p=1 holds k0+32; prefetch k0+64 into p=0 while computing
      const int kn = k0 + 64;
      const bool more = kn < DDIM;
      uint4 a0, a1, b0, b1;
      if (more) {
        a0 = *(const uint4*)(Arow0 + kn);
        a1 = *(const uint4*)(Arow1 + kn);
        b0 = *(const uint4*)(Brow0 + kn);
        b1 = *(const uint4*)(Brow1 + kn);
      }
      wave_mma_step(sA[1], sB[1], wm, wn, lane, acc);
      if (more) {
        *(uint4*)&sA[0][sOffLo] = a0;
        *(uint4*)&sA[0][sOffHi] = a1;
        *(uint4*)&sB[0][sOffLo] = b0;
        *(uint4*)&sB[0][sOffHi] = b1;
      }
      __syncthreads();
    }
  }

  // Epilogue: bias + SiLU, store bf16 h by list position.
  const int mrow = (lane >> 4) * 8;
  const int ncol = lane & 15;
#pragma unroll
  for (int j = 0; j < 4; ++j) {
    const int gn = (int)blockIdx.x * 128 + wn * 64 + j * 16 + ncol;
    const float bias = bh[(size_t)e * HDIM + gn];
#pragma unroll
    for (int i = 0; i < 2; ++i) {
#pragma unroll
      for (int r = 0; r < 8; ++r) {
        const int gm = (int)blockIdx.y * 128 + wm * 32 + i * 16 + mrow + r;
        if (gm < cnt) {
          h[(size_t)gm * HDIM + gn] = f32_to_bf16_rte(silu_fast(acc[i][j][r] + bias));
        }
      }
    }
  }
}

// out[tok] += gate * (h @ W2[e]^T + b2[e])
__global__ __launch_bounds__(256)
void moe_ffn2_kernel(const unsigned short* __restrict__ h,
                     const unsigned short* __restrict__ W2b,
                     const float* __restrict__ b2,
                     const float* __restrict__ gates,
                     const int* __restrict__ counts,
                     const int* __restrict__ lists,
                     float* __restrict__ out,
                     int e) {
  const int cnt = counts[e];
  if ((int)(blockIdx.y * 128) >= cnt) return;

  __shared__ unsigned short sA[2][128 * ST];
  __shared__ unsigned short sB[2][128 * ST];
  __shared__ int sTok[128];

  const int tid  = threadIdx.x;
  const int lane = tid & 31;
  const int wid  = tid >> 5;
  const int wm   = wid & 3;
  const int wn   = wid >> 2;

  const int* lst = lists + e * T_TOK;
  {
    int idx0 = (int)blockIdx.y * 128 + (tid & 127);
    if (idx0 >= cnt) idx0 = cnt - 1;
    if (tid < 128) sTok[tid] = lst[idx0];
  }

  const int arow = tid >> 2;
  const int aseg = (tid & 3) * 8;
  const int sOffLo = arow * ST + aseg;
  const int sOffHi = (arow + 64) * ST + aseg;
  // h rows are list positions directly (compacted), no gather on loads.
  const unsigned short* Arow0 = h + (size_t)((int)blockIdx.y * 128 + arow) * HDIM + aseg;
  const unsigned short* Arow1 = Arow0 + (size_t)64 * HDIM;
  const unsigned short* Brow0 = W2b + (size_t)e * ODIM * HDIM
                              + (size_t)((int)blockIdx.x * 128 + arow) * HDIM + aseg;
  const unsigned short* Brow1 = Brow0 + (size_t)64 * HDIM;

  v8f acc[2][4];
#pragma unroll
  for (int i = 0; i < 2; ++i)
#pragma unroll
    for (int j = 0; j < 4; ++j)
#pragma unroll
      for (int r = 0; r < 8; ++r) acc[i][j][r] = 0.f;

  {
    uint4 a0 = *(const uint4*)(Arow0);
    uint4 a1 = *(const uint4*)(Arow1);
    uint4 b0 = *(const uint4*)(Brow0);
    uint4 b1 = *(const uint4*)(Brow1);
    *(uint4*)&sA[0][sOffLo] = a0;
    *(uint4*)&sA[0][sOffHi] = a1;
    *(uint4*)&sB[0][sOffLo] = b0;
    *(uint4*)&sB[0][sOffHi] = b1;
  }
  __syncthreads();

  for (int k0 = 0; k0 < HDIM; k0 += 64) {
    {
      const int kn = k0 + 32;
      uint4 a0 = *(const uint4*)(Arow0 + kn);
      uint4 a1 = *(const uint4*)(Arow1 + kn);
      uint4 b0 = *(const uint4*)(Brow0 + kn);
      uint4 b1 = *(const uint4*)(Brow1 + kn);
      wave_mma_step(sA[0], sB[0], wm, wn, lane, acc);
      *(uint4*)&sA[1][sOffLo] = a0;
      *(uint4*)&sA[1][sOffHi] = a1;
      *(uint4*)&sB[1][sOffLo] = b0;
      *(uint4*)&sB[1][sOffHi] = b1;
      __syncthreads();
    }
    {
      const int kn = k0 + 64;
      const bool more = kn < HDIM;
      uint4 a0, a1, b0, b1;
      if (more) {
        a0 = *(const uint4*)(Arow0 + kn);
        a1 = *(const uint4*)(Arow1 + kn);
        b0 = *(const uint4*)(Brow0 + kn);
        b1 = *(const uint4*)(Brow1 + kn);
      }
      wave_mma_step(sA[1], sB[1], wm, wn, lane, acc);
      if (more) {
        *(uint4*)&sA[0][sOffLo] = a0;
        *(uint4*)&sA[0][sOffHi] = a1;
        *(uint4*)&sB[0][sOffLo] = b0;
        *(uint4*)&sB[0][sOffHi] = b1;
      }
      __syncthreads();
    }
  }

  const int mrow = (lane >> 4) * 8;
  const int ncol = lane & 15;
#pragma unroll
  for (int j = 0; j < 4; ++j) {
    const int gn = (int)blockIdx.x * 128 + wn * 64 + j * 16 + ncol;
    const float bias = b2[(size_t)e * ODIM + gn];
#pragma unroll
    for (int i = 0; i < 2; ++i) {
#pragma unroll
      for (int r = 0; r < 8; ++r) {
        const int pos = (int)blockIdx.y * 128 + wm * 32 + i * 16 + mrow + r;
        if (pos < cnt) {
          const int tok = sTok[wm * 32 + i * 16 + mrow + r];
          const float g = gates[(size_t)tok * NEXP + e];
          out[(size_t)tok * ODIM + gn] += g * (acc[i][j][r] + bias);
        }
      }
    }
  }
}

// ---------------------------------------------------------------- launch
extern "C" void kernel_launch(void* const* d_in, const int* in_sizes, int n_in,
                              void* d_out, int out_size, void* d_ws, size_t ws_size,
                              hipStream_t stream) {
  const float* x  = (const float*)d_in[0];  // [T, D]
  const float* Wg = (const float*)d_in[1];  // [E, D]
  const float* bg = (const float*)d_in[2];  // [E]
  const float* Wh = (const float*)d_in[3];  // [E, H, D]
  const float* bh = (const float*)d_in[4];  // [E, H]
  const float* W2 = (const float*)d_in[5];  // [E, O, H]
  const float* b2 = (const float*)d_in[6];  // [E, O]
  float* out = (float*)d_out;               // [T, O]

  char* ws = (char*)d_ws;
  size_t off = 0;
  float* gates = (float*)(ws + off);          off += (size_t)T_TOK * NEXP * 4;
  int*   counts = (int*)(ws + off);           off += 256;
  int*   lists = (int*)(ws + off);            off += (size_t)NEXP * T_TOK * 4;
  unsigned short* xb  = (unsigned short*)(ws + off); off += (size_t)T_TOK * DDIM * 2;
  unsigned short* Whb = (unsigned short*)(ws + off); off += (size_t)NEXP * HDIM * DDIM * 2;
  unsigned short* W2b = (unsigned short*)(ws + off); off += (size_t)NEXP * ODIM * HDIM * 2;
  unsigned short* hbuf = (unsigned short*)(ws + off); off += (size_t)T_TOK * HDIM * 2;
  (void)ws_size; (void)in_sizes; (void)n_in; (void)out_size;

  zero_i32_kernel<<<1, 32, 0, stream>>>(counts, NEXP);
  zero_f32_kernel<<<2048, 256, 0, stream>>>(out, (size_t)T_TOK * ODIM / 4);

  moe_gate_kernel<<<T_TOK / 8, 256, 0, stream>>>(x, Wg, bg, gates, counts, lists);

  cvt_f32_bf16_kernel<<<4096, 256, 0, stream>>>(x,  xb,  (size_t)T_TOK * DDIM / 4);
  cvt_f32_bf16_kernel<<<8192, 256, 0, stream>>>(Wh, Whb, (size_t)NEXP * HDIM * DDIM / 4);
  cvt_f32_bf16_kernel<<<8192, 256, 0, stream>>>(W2, W2b, (size_t)NEXP * ODIM * HDIM / 4);

  for (int e = 0; e < NEXP; ++e) {
    dim3 g1(HDIM / 128, T_TOK / 128);
    moe_ffn1_kernel<<<g1, 256, 0, stream>>>(xb, Whb, bh, counts, lists, hbuf, e);
    dim3 g2(ODIM / 128, T_TOK / 128);
    moe_ffn2_kernel<<<g2, 256, 0, stream>>>(hbuf, W2b, b2, gates, counts, lists, out, e);
  }
}
